// DecoderHierarchical_82394652606982
// MI455X (gfx1250) — compile-verified, matrix operand order checked by reading, not run on hardware
//
#include <hip/hip_runtime.h>
#include <hip/hip_bf16.h>

// ---------------- model constants ----------------
#define BATCH  512
#define LATENTD 512
#define NSUB   16
#define SUBSEQ 16
#define TSTEPS 256
#define CH     1024     // cond_hidden
#define CO     512      // cond_output_dim
#define DH     1024     // dec_hidden
#define INPD   128
#define NCLS   2

typedef __attribute__((ext_vector_type(16))) __bf16 v16bf;
typedef __attribute__((ext_vector_type(8)))  float  v8f;

__device__ __forceinline__ unsigned short f2bf(float f) {
    unsigned int u = __float_as_uint(f);
    u += 0x7fffu + ((u >> 16) & 1u);          // round-to-nearest-even
    return (unsigned short)(u >> 16);
}
__device__ __forceinline__ float sigmoidf(float x) { return 1.0f / (1.0f + expf(-x)); }

// ---------------- f32 -> bf16 conversion ----------------
__global__ void k_cvt_bf16(const float* __restrict__ src, unsigned short* __restrict__ dst, int n) {
    int i = blockIdx.x * 256 + threadIdx.x;
    if (i < n) dst[i] = f2bf(src[i]);
}
__global__ void k_fill_zero_bf(unsigned short* __restrict__ p, int n) {
    int i = blockIdx.x * 256 + threadIdx.x;
    if (i < n) p[i] = 0;
}

// ---------------- WMMA GEMM: C = act(A(MxK) * W(NxK)^T + bias [+ Cinit]) ----------------
// A, W bf16 row-major (lda/ldw = row stride in elements).
// Wave tile: 16(M) x 64(N) via 4 accumulators sharing one A fragment per K-step.
// Block: 8 waves = 4 along N x 2 along M -> 256(N) x 32(M) macro-tile.
// Requires M%32==0, N%256==0, K%32==0 (true for every GEMM in this model) -> no tails,
// EXEC all-ones as WMMA requires.
__global__ __launch_bounds__(256)
void k_wmma_gemm(const unsigned short* __restrict__ A, int lda,
                 const unsigned short* __restrict__ W, int ldw,
                 const float* __restrict__ bias,
                 const float* __restrict__ Cinit,          // row stride N, may be null
                 float* __restrict__ Cout,                 // row stride N, may be null
                 unsigned short* __restrict__ Cout_bf,     // bf16 mirror, may be null
                 int N, int K, int act)                    // act: 0=none, 1=tanh
{
    const int lane = threadIdx.x & 31;
    const int wave = threadIdx.x >> 5;
    const int wn   = wave & 3;                // wave position along N
    const int wm   = wave >> 2;               // wave position along M
    const int n0   = blockIdx.x * 256 + wn * 64;
    const int m0   = blockIdx.y * 32 + wm * 16;
    const int cn   = lane & 15;
    const int rsel = (lane >> 4) << 3;        // 0 or 8: C row offset AND K-half offset

    v8f acc[4];
#pragma unroll
    for (int j = 0; j < 4; ++j) {
        const int col = n0 + j * 16 + cn;
        const float binit = bias ? bias[col] : 0.0f;
#pragma unroll
        for (int r = 0; r < 8; ++r) {
            float v = binit;
            if (Cinit) v += Cinit[(m0 + r + rsel) * N + col];
            acc[j][r] = v;
        }
    }

    const unsigned short* arow = A + (m0 + cn) * lda + rsel;
    const unsigned short* wrow0 = W + (n0 + cn) * ldw + rsel;           // j=0
    const unsigned short* wrow1 = W + (n0 + 16 + cn) * ldw + rsel;      // j=1
    const unsigned short* wrow2 = W + (n0 + 32 + cn) * ldw + rsel;      // j=2
    const unsigned short* wrow3 = W + (n0 + 48 + cn) * ldw + rsel;      // j=3

    for (int k0 = 0; k0 < K; k0 += 32) {
        union { uint4 u[2]; v16bf v; } fa, fb0, fb1, fb2, fb3;
        fa.u[0]  = *(const uint4*)(arow  + k0);
        fa.u[1]  = *(const uint4*)(arow  + k0 + 16);
        fb0.u[0] = *(const uint4*)(wrow0 + k0);
        fb0.u[1] = *(const uint4*)(wrow0 + k0 + 16);
        fb1.u[0] = *(const uint4*)(wrow1 + k0);
        fb1.u[1] = *(const uint4*)(wrow1 + k0 + 16);
        fb2.u[0] = *(const uint4*)(wrow2 + k0);
        fb2.u[1] = *(const uint4*)(wrow2 + k0 + 16);
        fb3.u[0] = *(const uint4*)(wrow3 + k0);
        fb3.u[1] = *(const uint4*)(wrow3 + k0 + 16);
        acc[0] = __builtin_amdgcn_wmma_f32_16x16x32_bf16(
                     false, fa.v, false, fb0.v, (short)0, acc[0], false, false);
        acc[1] = __builtin_amdgcn_wmma_f32_16x16x32_bf16(
                     false, fa.v, false, fb1.v, (short)0, acc[1], false, false);
        acc[2] = __builtin_amdgcn_wmma_f32_16x16x32_bf16(
                     false, fa.v, false, fb2.v, (short)0, acc[2], false, false);
        acc[3] = __builtin_amdgcn_wmma_f32_16x16x32_bf16(
                     false, fa.v, false, fb3.v, (short)0, acc[3], false, false);
    }

#pragma unroll
    for (int j = 0; j < 4; ++j) {
        const int col = n0 + j * 16 + cn;
#pragma unroll
        for (int r = 0; r < 8; ++r) {
            float v = acc[j][r];
            if (act == 1) v = tanhf(v);
            int idx = (m0 + r + rsel) * N + col;
            if (Cout)    Cout[idx]    = v;
            if (Cout_bf) Cout_bf[idx] = f2bf(v);
        }
    }
}

static inline dim3 gemm_grid(int M, int N) { return dim3(N / 256, M / 32); }

// ---------------- LSTM cell pointwise (gates order i,f,g,o; chunks of CH) ----------------
__global__ void k_lstm_pointwise(const float* __restrict__ gates,
                                 const float* __restrict__ c_in,
                                 float* __restrict__ c_out,
                                 unsigned short* __restrict__ h_bf,        // contiguous (B,CH)
                                 unsigned short* __restrict__ hseq_bf,     // slot pointer, row stride seq_ld
                                 int seq_ld)
{
    int i = blockIdx.x * 256 + threadIdx.x;
    if (i >= BATCH * CH) return;
    int b = i >> 10, j = i & 1023;
    const float* g = gates + b * (4 * CH);
    float ig = sigmoidf(g[j]);
    float fg = sigmoidf(g[CH + j]);
    float gg = tanhf   (g[2 * CH + j]);
    float og = sigmoidf(g[3 * CH + j]);
    float c  = fg * c_in[i] + ig * gg;
    c_out[i] = c;
    unsigned short hb = f2bf(og * tanhf(c));
    h_bf[i] = hb;
    hseq_bf[b * seq_ld + j] = hb;
}

// ---------------- h0m = mean over NL=2 of raw-reshaped tanh(fid) output ----------------
__global__ void k_mean_h0(const float* __restrict__ h0s, float* __restrict__ h0m, int n) {
    int i = blockIdx.x * 256 + threadIdx.x;
    if (i < n) h0m[i] = 0.5f * (h0s[i] + h0s[n + i]);
}

// ---------------- decoder h reset at subsequence start ----------------
__global__ void k_reset_h(const float* __restrict__ h0m, int s,
                          float* __restrict__ h, unsigned short* __restrict__ h_bf) {
    int i = blockIdx.x * 256 + threadIdx.x;
    if (i >= BATCH * DH) return;
    int b = i >> 10, j = i & 1023;
    float v = h0m[b * (NSUB * DH) + s * DH + j];
    h[i] = v;
    h_bf[i] = f2bf(v);
}

// ---------------- GRU cell pointwise (gates order r,z,n; chunks of DH) ----------------
__global__ void k_gru_pointwise(const float* __restrict__ gi, const float* __restrict__ gh,
                                float* __restrict__ h, unsigned short* __restrict__ h_bf) {
    int i = blockIdx.x * 256 + threadIdx.x;
    if (i >= BATCH * DH) return;
    int b = i >> 10, j = i & 1023;
    const float* pi = gi + b * (3 * DH);
    const float* ph = gh + b * (3 * DH);
    float r = sigmoidf(pi[j] + ph[j]);
    float z = sigmoidf(pi[DH + j] + ph[DH + j]);
    float n = tanhf(pi[2 * DH + j] + r * ph[2 * DH + j]);
    float hn = (1.0f - z) * n + z * h[i];
    h[i] = hn;
    h_bf[i] = f2bf(hn);
}

// ---------------- log_softmax over NCLS=2, argmax -> one-hot token, write d_out ----------------
__global__ void k_emit(const float* __restrict__ logits, float* __restrict__ out,
                       unsigned short* __restrict__ token_bf, int t) {
    int i = blockIdx.x * 256 + threadIdx.x;
    if (i >= BATCH * INPD) return;
    int b = i >> 7, j = i & 127;
    float x0 = logits[b * 256 + j];
    float x1 = logits[b * 256 + 128 + j];
    float m  = fmaxf(x0, x1);
    float lse = m + logf(expf(x0 - m) + expf(x1 - m));
    float* o = out + (size_t)b * TSTEPS * 256 + (size_t)t * 256;
    o[j]       = x0 - lse;
    o[128 + j] = x1 - lse;
    int idx = (x1 > x0) ? 1 : 0;                  // argmax, ties -> class 0
    token_bf[b * 256 + 2 * j]     = (idx == 0) ? 0x3F80 : 0;   // bf16 1.0
    token_bf[b * 256 + 2 * j + 1] = (idx == 1) ? 0x3F80 : 0;
}

// =====================================================================================
extern "C" void kernel_launch(void* const* d_in, const int* in_sizes, int n_in,
                              void* d_out, int out_size, void* d_ws, size_t ws_size,
                              hipStream_t stream) {
    (void)in_sizes; (void)n_in; (void)out_size; (void)ws_size;

    const float* latent  = (const float*)d_in[0];
    const float* fic_w   = (const float*)d_in[1];
    const float* fic_b   = (const float*)d_in[2];
    const float* l0_wih  = (const float*)d_in[3];
    const float* l0_whh  = (const float*)d_in[4];
    const float* l0_bih  = (const float*)d_in[5];
    const float* l0_bhh  = (const float*)d_in[6];
    const float* l1_wih  = (const float*)d_in[7];
    const float* l1_whh  = (const float*)d_in[8];
    const float* l1_bih  = (const float*)d_in[9];
    const float* l1_bhh  = (const float*)d_in[10];
    const float* co_w    = (const float*)d_in[11];
    const float* co_b    = (const float*)d_in[12];
    const float* fid_w   = (const float*)d_in[13];
    const float* fid_b   = (const float*)d_in[14];
    const float* g_wih   = (const float*)d_in[15];
    const float* g_whh   = (const float*)d_in[16];
    const float* g_bih   = (const float*)d_in[17];
    const float* g_bhh   = (const float*)d_in[18];
    const float* do_w    = (const float*)d_in[19];
    const float* do_b    = (const float*)d_in[20];
    float* out = (float*)d_out;

    // ---- workspace bump allocator ----
    char* base = (char*)d_ws;
    size_t off = 0;
    auto alloc = [&](size_t elems, size_t esz) -> void* {
        void* p = base + off;
        off = (off + elems * esz + 255) & ~(size_t)255;
        return p;
    };

    // bf16 weight mirrors
    unsigned short* fic_bf    = (unsigned short*)alloc((size_t)2048 * 512, 2);
    unsigned short* l0_wih_bf = (unsigned short*)alloc((size_t)4096 * 32, 2);
    unsigned short* l0_whh_bf = (unsigned short*)alloc((size_t)4096 * 1024, 2);
    unsigned short* l1_wih_bf = (unsigned short*)alloc((size_t)4096 * 1024, 2);
    unsigned short* l1_whh_bf = (unsigned short*)alloc((size_t)4096 * 1024, 2);
    unsigned short* co_bf     = (unsigned short*)alloc((size_t)512 * 1024, 2);
    unsigned short* fid_bf    = (unsigned short*)alloc((size_t)2048 * 512, 2);
    unsigned short* g_wih_bf  = (unsigned short*)alloc((size_t)3072 * 768, 2);
    unsigned short* g_whh_bf  = (unsigned short*)alloc((size_t)3072 * 1024, 2);
    unsigned short* do_bf     = (unsigned short*)alloc((size_t)256 * 1024, 2);
    // activations
    unsigned short* latent_bf = (unsigned short*)alloc((size_t)BATCH * LATENTD, 2);
    float*          h0c_f32   = (float*)         alloc((size_t)BATCH * 2 * CH, 4);
    unsigned short* h0c_bf    = (unsigned short*)alloc((size_t)BATCH * 2 * CH, 2);
    float*          gates     = (float*)         alloc((size_t)BATCH * 4 * CH, 4);
    float*          c_buf     = (float*)         alloc((size_t)BATCH * CH, 4);
    unsigned short* hl_bf     = (unsigned short*)alloc((size_t)BATCH * CH, 2);
    unsigned short* xseq0_bf  = (unsigned short*)alloc((size_t)BATCH * NSUB * CH, 2);
    unsigned short* xseq1_bf  = (unsigned short*)alloc((size_t)BATCH * NSUB * CH, 2);
    float*          emb_f32   = (float*)         alloc((size_t)BATCH * NSUB * CO, 4);
    unsigned short* emb_bf    = (unsigned short*)alloc((size_t)BATCH * NSUB * CO, 2);
    float*          h0m       = (float*)         alloc((size_t)BATCH * NSUB * DH, 4);
    size_t h0s_off = off;                       // h0s region; aliased later by decoder scratch
    float*          h0s       = (float*)         alloc((size_t)BATCH * NSUB * 2 * DH, 4);

    auto CVT = [&](const float* s, unsigned short* d, int n) {
        k_cvt_bf16<<<(n + 255) / 256, 256, 0, stream>>>(s, d, n);
    };
    CVT(fic_w,  fic_bf,    2048 * 512);
    CVT(l0_wih, l0_wih_bf, 4096 * 32);
    CVT(l0_whh, l0_whh_bf, 4096 * 1024);
    CVT(l1_wih, l1_wih_bf, 4096 * 1024);
    CVT(l1_whh, l1_whh_bf, 4096 * 1024);
    CVT(co_w,   co_bf,     512 * 1024);
    CVT(fid_w,  fid_bf,    2048 * 512);
    CVT(g_wih,  g_wih_bf,  3072 * 768);
    CVT(g_whh,  g_whh_bf,  3072 * 1024);
    CVT(do_w,   do_bf,     256 * 1024);
    CVT(latent, latent_bf, BATCH * LATENTD);

    // ---- h0_cond = tanh(latent @ fic_w.T + fic_b)  (M=512, N=2048, K=512) ----
    k_wmma_gemm<<<gemm_grid(BATCH, 2048), 256, 0, stream>>>(
        latent_bf, LATENTD, fic_bf, LATENTD, fic_b, nullptr,
        h0c_f32, h0c_bf, 2048, 512, 1);

    // ---- conductor LSTM: layer 0 then layer 1, 16 steps each ----
    const int PW = (BATCH * CH + 255) / 256;
    for (int layer = 0; layer < 2; ++layer) {
        const unsigned short* wih  = layer ? l1_wih_bf : l0_wih_bf;
        const unsigned short* whh  = layer ? l1_whh_bf : l0_whh_bf;
        const float* bih = layer ? l1_bih : l0_bih;
        const float* bhh = layer ? l1_bhh : l0_bhh;
        const int kin = layer ? CH : (LATENTD / NSUB);
        unsigned short* xout = layer ? xseq1_bf : xseq0_bf;

        const float* c_in = h0c_f32 + (size_t)layer * BATCH * CH;
        const unsigned short* h_prev = h0c_bf + (size_t)layer * BATCH * CH;
        for (int s = 0; s < NSUB; ++s) {
            const unsigned short* xt = layer ? (xseq0_bf + s * CH) : (latent_bf + s * (LATENTD / NSUB));
            const int ldx = layer ? (NSUB * CH) : LATENTD;
            // gates = xt @ wih.T + bih
            k_wmma_gemm<<<gemm_grid(BATCH, 4096), 256, 0, stream>>>(
                xt, ldx, wih, kin, bih, nullptr, gates, nullptr, 4096, kin, 0);
            // gates += h @ whh.T + bhh
            k_wmma_gemm<<<gemm_grid(BATCH, 4096), 256, 0, stream>>>(
                h_prev, CH, whh, CH, bhh, gates, gates, nullptr, 4096, CH, 0);
            k_lstm_pointwise<<<PW, 256, 0, stream>>>(gates, c_in, c_buf,
                                                     hl_bf, xout + s * CH, NSUB * CH);
            c_in = c_buf;
            h_prev = hl_bf;
        }
    }

    // ---- emb = x @ co_w.T + co_b  (M = B*NSUB = 8192, N=512, K=1024) ----
    k_wmma_gemm<<<gemm_grid(BATCH * NSUB, 512), 256, 0, stream>>>(
        xseq1_bf, CH, co_bf, CH, co_b, nullptr, emb_f32, emb_bf, 512, CH, 0);

    // ---- h0s = tanh(emb @ fid_w.T + fid_b)  (M=8192, N=2048, K=512) ----
    k_wmma_gemm<<<gemm_grid(BATCH * NSUB, 2048), 256, 0, stream>>>(
        emb_bf, CO, fid_bf, CO, fid_b, nullptr, h0s, nullptr, 2048, CO, 1);
    // h0m = mean over NL (raw .view semantics)
    {
        int n = BATCH * NSUB * DH;
        k_mean_h0<<<(n + 255) / 256, 256, 0, stream>>>(h0s, h0m, n);
    }

    // ---- decoder scratch aliases dead h0s region ----
    off = h0s_off;
    float*          gie      = (float*)         alloc((size_t)BATCH * 3 * DH, 4);
    float*          gi       = (float*)         alloc((size_t)BATCH * 3 * DH, 4);
    float*          gh       = (float*)         alloc((size_t)BATCH * 3 * DH, 4);
    float*          logits   = (float*)         alloc((size_t)BATCH * 256, 4);
    float*          hdec     = (float*)         alloc((size_t)BATCH * DH, 4);
    unsigned short* hdec_bf  = (unsigned short*)alloc((size_t)BATCH * DH, 2);
    unsigned short* token_bf = (unsigned short*)alloc((size_t)BATCH * INPD * NCLS, 2);

    k_fill_zero_bf<<<(BATCH * INPD * NCLS + 255) / 256, 256, 0, stream>>>(
        token_bf, BATCH * INPD * NCLS);

    // ---- 256-step GRU decoder with argmax feedback ----
    for (int t = 0; t < TSTEPS; ++t) {
        const int s = t >> 4;
        if ((t & (SUBSEQ - 1)) == 0) {
            // h <- h0m[:, s, :]
            k_reset_h<<<(BATCH * DH + 255) / 256, 256, 0, stream>>>(h0m, s, hdec, hdec_bf);
            // gie = e @ Wih_e.T + bih  (constant for the next 16 steps; K=CO=512)
            k_wmma_gemm<<<gemm_grid(BATCH, 3072), 256, 0, stream>>>(
                emb_bf + s * CO, NSUB * CO, g_wih_bf + (INPD * NCLS), INPD * NCLS + CO,
                g_bih, nullptr, gie, nullptr, 3072, CO, 0);
        }
        // gi = token @ Wih_t.T + gie  (K = 256)
        k_wmma_gemm<<<gemm_grid(BATCH, 3072), 256, 0, stream>>>(
            token_bf, INPD * NCLS, g_wih_bf, INPD * NCLS + CO,
            nullptr, gie, gi, nullptr, 3072, INPD * NCLS, 0);
        // gh = h @ Whh.T + bhh  (K = 1024)
        k_wmma_gemm<<<gemm_grid(BATCH, 3072), 256, 0, stream>>>(
            hdec_bf, DH, g_whh_bf, DH, g_bhh, nullptr, gh, nullptr, 3072, DH, 0);
        // GRU pointwise update
        k_gru_pointwise<<<(BATCH * DH + 255) / 256, 256, 0, stream>>>(gi, gh, hdec, hdec_bf);
        // logits = h @ do_w.T + do_b  (N=256, K=1024)
        k_wmma_gemm<<<gemm_grid(BATCH, 256), 256, 0, stream>>>(
            hdec_bf, DH, do_bf, DH, do_b, nullptr, logits, nullptr, 256, DH, 0);
        // log_softmax + argmax one-hot feedback + output write
        k_emit<<<(BATCH * INPD + 255) / 256, 256, 0, stream>>>(logits, out, token_bf, t);
    }
}
